// BiDAF_14087492730880
// MI455X (gfx1250) — matrix operand order repaired and checked by reading, not compiled, not run on hardware
//
#include <hip/hip_runtime.h>

#define BB 64
#define CC 512
#define QQ 64
#define HH 256
#define NW 8     // waves per block (256 threads, wave32)

typedef __attribute__((ext_vector_type(16))) __bf16 v16bf;
typedef __attribute__((ext_vector_type(8)))  float  v8f;

struct alignas(32) SMem {
  __bf16 q_bf [QQ * HH];       // 32768 B : q row-major (B operand for S)
  __bf16 qT_bf[HH * QQ];       // 32768 B : q transposed (B operand for c2q)
  float  wm   [HH];            //  1024 B : w_m
  float  cdot [CC];            //  2048 B : c @ w_c
  float  qdot [QQ];            //   256 B : q @ w_q
  float  rmax [CC];            //  2048 B : max_j S  -> later exp weights
  float  red  [256];           //  1024 B
  float  redB [256];           //  1024 B
  float  maxc2q [HH];          //  1024 B : max_i c2q
  float  maxprod[HH];          //  1024 B : max_i (c * c2q)
  float  finalv[4 * HH];       //  4096 B : pooled features
  float  sstrip[NW][16 * QQ];  // 32768 B : per-wave S strip (f32)
  __bf16 pstrip[NW][16 * QQ];  // 16384 B : per-wave softmax probs (bf16)
};                             // ~124 KB total

__device__ __forceinline__ void lds_atomic_max_f32(float* p, float v) {
  int old = __float_as_int(*p);
  while (v > __int_as_float(old)) {
    int assumed = old;
    old = atomicCAS((int*)p, assumed, __float_as_int(v));
    if (old == assumed) break;
  }
}

__global__ __launch_bounds__(256)
void bidaf_kernel(const float* __restrict__ Cg, const float* __restrict__ Qg,
                  const float* __restrict__ wsim, const float* __restrict__ wlab,
                  const float* __restrict__ blab, float* __restrict__ out)
{
  __shared__ SMem sm;
  const int b    = blockIdx.x;
  const int t    = threadIdx.x;
  const int lane = t & 31;
  const int wv   = t >> 5;
  const int hi   = lane >> 4;   // half-wave id (0/1)
  const int ln   = lane & 15;
  const float* cb = Cg + (size_t)b * CC * HH;
  const float* qb = Qg + (size_t)b * QQ * HH;

  // ---------------- Phase 0: stage q (bf16, both layouts), w_m, dots, inits -----
  for (int idx = t; idx < QQ * HH; idx += 256) {
    const int j = idx >> 8, d = idx & (HH - 1);
    const float v = qb[idx];
    const __bf16 bv = (__bf16)v;
    sm.q_bf[idx]       = bv;
    sm.qT_bf[d*QQ + j] = bv;
  }
  if (t < HH) {
    sm.wm[t]      = wsim[2*HH + t];
    sm.maxc2q[t]  = -3.4e38f;
    sm.maxprod[t] = -3.4e38f;
  }
  if (t < QQ) {
    float acc = 0.f;
    const float* qr = qb + t * HH;
    for (int d = 0; d < HH; ++d) acc += qr[d] * wsim[HH + d];
    sm.qdot[t] = acc;
  }
  for (int i = t; i < CC; i += 256) {
    float acc = 0.f;
    const float* cr = cb + (size_t)i * HH;
    for (int d = 0; d < HH; ++d) acc += cr[d] * wsim[d];
    sm.cdot[i] = acc;
  }
  __syncthreads();

  // ---------------- Phase A: per row-tile fused S -> softmax -> c2q -------------
  float* sstrip = sm.sstrip[wv];
  __bf16* pstrip = sm.pstrip[wv];
  const int akb = hi * 16;  // A-operand K base within a 32-chunk

  for (int rt = wv; rt < CC/16; rt += NW) {
    const int rowbase = rt * 16;
    const float* crow = cb + (size_t)(rowbase + ln) * HH;   // this lane's A row
    // warm next tile of c
    __builtin_prefetch(cb + (size_t)(rowbase + 16*NW + ln) * HH, 0, 1);

    // ---- S tiles: (c * w_m) @ q^T, K = 256 ----
    v8f sacc[4] = {};
    for (int kb = 0; kb < 8; ++kb) {
      v16bf A;
      const int k0 = kb*32 + akb;
#pragma unroll
      for (int e = 0; e < 8; ++e) {
        A[e]     = (__bf16)(crow[k0 + e]      * sm.wm[k0 + e]);
        A[e + 8] = (__bf16)(crow[k0 + 8 + e] * sm.wm[k0 + 8 + e]);
      }
#pragma unroll
      for (int ct = 0; ct < 4; ++ct) {
        const int j = ct*16 + ln;
        const v16bf Bm = *(const v16bf*)&sm.q_bf[j*HH + kb*32 + hi*16];
        sacc[ct] = __builtin_amdgcn_wmma_f32_16x16x32_bf16(
            false, A, false, Bm, (short)0, sacc[ct], false, false);
      }
    }
    // add row/col bias terms, spill strip to LDS (D layout: M = r + 8*hi, N = ln)
#pragma unroll
    for (int ct = 0; ct < 4; ++ct) {
      const int j = ct*16 + ln;
      const float qd = sm.qdot[j];
#pragma unroll
      for (int r = 0; r < 8; ++r) {
        const int m = r + hi*8;
        sstrip[m*QQ + j] = sacc[ct][r] + sm.cdot[rowbase + m] + qd;
      }
    }

    // ---- row softmax over j (16 rows -> lanes 0..15) ----
    if (lane < 16) {
      const int m = lane;
      float mx = -3.4e38f;
      for (int j = 0; j < QQ; ++j) mx = fmaxf(mx, sstrip[m*QQ + j]);
      float sum = 0.f;
      for (int j = 0; j < QQ; ++j) {
        const float e = __expf(sstrip[m*QQ + j] - mx);
        sstrip[m*QQ + j] = e;
        sum += e;
      }
      const float inv = 1.f / sum;
      for (int j = 0; j < QQ; ++j)
        pstrip[m*QQ + j] = (__bf16)(sstrip[m*QQ + j] * inv);
      sm.rmax[rowbase + m] = mx;
    }

    // ---- c2q tiles: P @ q, K = 64 ----
    v16bf A2[2];
#pragma unroll
    for (int kb = 0; kb < 2; ++kb) {
      const int k0 = kb*32 + akb;
#pragma unroll
      for (int e = 0; e < 8; ++e) {
        A2[kb][e]     = pstrip[ln*QQ + k0 + e];
        A2[kb][e + 8] = pstrip[ln*QQ + k0 + 8 + e];
      }
    }
    for (int dt = 0; dt < 16; ++dt) {
      v8f acc2 = {};
#pragma unroll
      for (int kb = 0; kb < 2; ++kb) {
        const int dcol = dt*16 + ln;
        const v16bf B2 = *(const v16bf*)&sm.qT_bf[dcol*QQ + kb*32 + hi*16];
        acc2 = __builtin_amdgcn_wmma_f32_16x16x32_bf16(
            false, A2[kb], false, B2, (short)0, acc2, false, false);
      }
      // column maxima of c2q and c*c2q for the max-pool
      const int d = dt*16 + ln;
      float mx1 = -3.4e38f, mx2 = -3.4e38f;
#pragma unroll
      for (int r = 0; r < 8; ++r) {
        const int i = rowbase + r + hi*8;
        const float cv = cb[(size_t)i*HH + d];
        const float v  = acc2[r];
        mx1 = fmaxf(mx1, v);
        mx2 = fmaxf(mx2, v * cv);
      }
      lds_atomic_max_f32(&sm.maxc2q[d],  mx1);
      lds_atomic_max_f32(&sm.maxprod[d], mx2);
    }
  }
  __syncthreads();

  // ---------------- Phase B: b_att softmax over C, q2c, pooled features ---------
  sm.red[t] = fmaxf(sm.rmax[t], sm.rmax[t + 256]);
  __syncthreads();
  for (int s = 128; s > 0; s >>= 1) {
    if (t < s) sm.red[t] = fmaxf(sm.red[t], sm.red[t + s]);
    __syncthreads();
  }
  const float Mv = sm.red[0];
  __syncthreads();
  {
    const float e0 = __expf(sm.rmax[t]       - Mv);
    const float e1 = __expf(sm.rmax[t + 256] - Mv);
    sm.rmax[t] = e0; sm.rmax[t + 256] = e1;
    sm.red[t] = e0 + e1;
  }
  __syncthreads();
  for (int s = 128; s > 0; s >>= 1) {
    if (t < s) sm.red[t] += sm.red[t + s];
    __syncthreads();
  }
  const float Z = sm.red[0];
  __syncthreads();

  {
    const int d = t;            // one thread per feature dim, coalesced over c
    float acc = 0.f, mxc = -3.4e38f, mnc = 3.4e38f;
    for (int i = 0; i < CC; ++i) {
      const float v = cb[(size_t)i*HH + d];
      acc += sm.rmax[i] * v;
      mxc = fmaxf(mxc, v);
      mnc = fminf(mnc, v);
    }
    const float q2 = acc / Z;   // q2c[d]
    sm.finalv[d]          = mxc;                        // max_i c
    sm.finalv[HH + d]     = sm.maxc2q[d];               // max_i c2q
    sm.finalv[2*HH + d]   = sm.maxprod[d];              // max_i c*c2q
    sm.finalv[3*HH + d]   = q2 * (q2 >= 0.f ? mxc : mnc); // max_i c*q2c
  }
  __syncthreads();

  // ---------------- Phase C: final @ w_label + b_label (deterministic tree) ----
  float p0 = 0.f, p1 = 0.f;
#pragma unroll
  for (int ss = 0; ss < 4; ++ss) {
    const int k = t + ss*256;
    const float fv = sm.finalv[k];
    p0 += fv * wlab[k*2 + 0];
    p1 += fv * wlab[k*2 + 1];
  }
  sm.red[t] = p0; sm.redB[t] = p1;
  __syncthreads();
  for (int s = 128; s > 0; s >>= 1) {
    if (t < s) { sm.red[t] += sm.red[t + s]; sm.redB[t] += sm.redB[t + s]; }
    __syncthreads();
  }
  if (t == 0) {
    out[b*2 + 0] = sm.red[0]  + blab[0];
    out[b*2 + 1] = sm.redB[0] + blab[1];
  }
}

extern "C" void kernel_launch(void* const* d_in, const int* in_sizes, int n_in,
                              void* d_out, int out_size, void* d_ws, size_t ws_size,
                              hipStream_t stream) {
  const float* c    = (const float*)d_in[0];
  const float* q    = (const float*)d_in[1];
  const float* wsim = (const float*)d_in[2];
  const float* wlab = (const float*)d_in[3];
  const float* blab = (const float*)d_in[4];
  float* out = (float*)d_out;
  bidaf_kernel<<<BB, 256, 0, stream>>>(c, q, wsim, wlab, blab, out);
}
